// Model_76304388980808
// MI455X (gfx1250) — compile-verified, matrix operand order checked by reading, not run on hardware
//
#include <hip/hip_runtime.h>
#include <hip/hip_bf16.h>
#include <math.h>

// ---------------------------------------------------------------------------
// LSTM LM forward on MI455X (gfx1250, wave32, WMMA bf16 -> f32 accumulate)
//
//   B=64, S=512, E=512, H=1024, V=32000, C=20
//
//   1) gather_embed : emb[x] -> xemb bf16 [B,S,E]
//   2) init_state   : zero a-buffers + grid-barrier counter
//   3) lstm_recurrent (PERSISTENT, 64 WGs x 128 thr, 193KB LDS):
//        - one hidden-column tile (j) per workgroup; gate weights converted
//          f32->bf16 into bank-conflict-free padded LDS ONCE, then resident
//          for all 512 steps (zero L2 weight traffic in the recurrence)
//        - fully unrolled 48-chunk K loop: W fragments via ds_load_b128 with
//          immediate offsets; A fragments via a 3-deep global prefetch
//          pipeline (2 loads/chunk, issued 3 chunks ahead)
//        - cell state in VGPRs, device-wide barrier per step
//   4) classifier_loss : a_fin @ Wv^T + bv -> log_softmax -> mean NLL
// ---------------------------------------------------------------------------

typedef __attribute__((ext_vector_type(16))) __bf16 v16bf;
typedef __attribute__((ext_vector_type(8)))  __bf16 v8bf;
typedef __attribute__((ext_vector_type(8)))  float  v8f;

#define B_   64
#define S_   512
#define E_   512
#define H_   1024
#define C_   20
#define KTOT (E_ + H_)   // 1536
#define NWG  64          // workgroups in persistent recurrent kernel
#define KCH  (KTOT / 32) // 48 K-chunks of 32
#define APIPE 3          // A-fragment prefetch depth (chunks)

// LDS weight tile layout (per workgroup): [gate][col][K] bf16, padded.
// col stride 3088B = 772 dwords; 772 % 64 == 4 -> 16 cols map to banks
// 0,4,8,...,60: conflict-free b128 reads across lanes.
#define CSTRIDE 3088
#define GSTRIDE (16 * CSTRIDE)        // 49,408 B per gate
#define LDSZ    (4 * GSTRIDE)         // 197,632 B per workgroup

// workspace layout (bytes)
#define XEMB_OFF 0u
#define XEMB_BYTES ((size_t)B_ * S_ * E_ * 2)                 // 33,554,432
#define ABUF_OFF (XEMB_OFF + XEMB_BYTES)
#define ABUF_BYTES ((size_t)2 * B_ * H_ * 2)                  // double buffer
#define AFIN_OFF (ABUF_OFF + ABUF_BYTES)
#define AFIN_BYTES ((size_t)B_ * H_ * 4)
#define CNT_OFF  (AFIN_OFF + AFIN_BYTES)

// Per-lane WMMA 16-bit fragment: two contiguous 16B chunks.
// elements 0..7  -> VGPR0..3 (K = (l>>4)*8 + 0..7)
// elements 8..15 -> VGPR4..7 (K = 16 + (l>>4)*8 + 0..7)
__device__ __forceinline__ v16bf mkfrag(v8bf lo, v8bf hi) {
  return __builtin_shufflevector(lo, hi, 0, 1, 2, 3, 4, 5, 6, 7,
                                 8, 9, 10, 11, 12, 13, 14, 15);
}
// global: uniform base + per-lane u32 byte offset + const imm
__device__ __forceinline__ v16bf ldfrag_g(const void* __restrict__ base,
                                          unsigned boff, int cofs) {
  const char* p = (const char*)base + boff;
  return mkfrag(*(const v8bf*)(p + cofs), *(const v8bf*)(p + cofs + 32));
}
// LDS: per-lane address VGPR + const imm (folds into 16-bit DS offset)
__device__ __forceinline__ v16bf ldfrag_l(const char* p, int cofs) {
  return mkfrag(*(const v8bf*)(p + cofs), *(const v8bf*)(p + cofs + 32));
}

__device__ __forceinline__ v8f splat8(float x) {
  v8f v = {x, x, x, x, x, x, x, x};
  return v;
}

__device__ __forceinline__ void wmma4(const v16bf a, const v16bf b0,
                                      const v16bf b1, const v16bf b2,
                                      const v16bf b3, v8f& f, v8f& i,
                                      v8f& c, v8f& o) {
  f = __builtin_amdgcn_wmma_f32_16x16x32_bf16(false, a, false, b0, (short)0, f,
                                              false, false);
  i = __builtin_amdgcn_wmma_f32_16x16x32_bf16(false, a, false, b1, (short)0, i,
                                              false, false);
  c = __builtin_amdgcn_wmma_f32_16x16x32_bf16(false, a, false, b2, (short)0, c,
                                              false, false);
  o = __builtin_amdgcn_wmma_f32_16x16x32_bf16(false, a, false, b3, (short)0, o,
                                              false, false);
}

// Branch-free fast nonlinearities (v_exp_f32 + v_rcp_f32, no EXEC churn).
__device__ __forceinline__ float fast_sigmoid(float x) {
  x = fminf(fmaxf(x, -30.0f), 30.0f);
  return __builtin_amdgcn_rcpf(1.0f + __expf(-x));
}
__device__ __forceinline__ float fast_tanh(float x) {
  x = fminf(fmaxf(x, -15.0f), 15.0f);
  const float e = __expf(-2.0f * x);
  return (1.0f - e) * __builtin_amdgcn_rcpf(1.0f + e);
}

// ------------------------------- prep kernels ------------------------------

__global__ void gather_embed(const int* __restrict__ x,
                             const float* __restrict__ emb,
                             __bf16* __restrict__ xemb) {
  const int bs = blockIdx.x;                    // b*S + s
  const int row = x[bs];
  const float* src = emb + (size_t)row * E_;
  __bf16* dst = xemb + (size_t)bs * E_;
  for (int e = threadIdx.x; e < E_; e += blockDim.x)
    dst[e] = (__bf16)src[e];
}

__global__ void init_state(__bf16* __restrict__ abuf, unsigned* __restrict__ cnt) {
  const int tot = 2 * B_ * H_;
  for (int i = blockIdx.x * blockDim.x + threadIdx.x; i < tot;
       i += gridDim.x * blockDim.x)
    abuf[i] = (__bf16)0.0f;
  if (blockIdx.x == 0 && threadIdx.x == 0) *cnt = 0u;
}

// --------------------------- persistent recurrence -------------------------

__global__ void __launch_bounds__(128) lstm_recurrent(
    const __bf16* __restrict__ xemb,   // [B,S,E]
    const float* __restrict__ Wf, const float* __restrict__ Wi,
    const float* __restrict__ Wc, const float* __restrict__ Wo,
    const float* __restrict__ bfp, const float* __restrict__ bip,
    const float* __restrict__ bcp, const float* __restrict__ bop,
    __bf16* abuf,                      // [2][B,H] double buffer
    float*  __restrict__ afin,         // [B,H] f32 final hidden
    unsigned* cnt) {
  extern __shared__ char smem[];      // LDSZ bytes: [gate][col][K] bf16 padded

  const int lane = threadIdx.x & 31;
  const int mt   = threadIdx.x >> 5;  // wave id = M tile (16 batch rows)
  const int j    = blockIdx.x;        // hidden-column tile 0..63
  const int l15  = lane & 15;
  const unsigned klB = (unsigned)((lane >> 4) << 4);  // per-lane K sub (bytes)
  const int arow = mt * 16 + l15;                     // A-fragment batch row

  const float* Wg[4] = {Wf, Wi, Wc, Wo};
  const float* bg[4] = {bfp, bip, bcp, bop};

  // ---- one-time: convert this block's weight tile f32 -> bf16 into LDS ----
  for (int w = threadIdx.x; w < 64 * KCH; w += blockDim.x) {
    const int row = w / KCH;          // 0..63 : gate*16 + col
    const int ch  = w % KCH;          // 32-element chunk
    const int g = row >> 4, c = row & 15;
    const float* src = Wg[g] + (size_t)(j * 16 + c) * KTOT + ch * 32;
    __bf16* dst = (__bf16*)(smem + g * GSTRIDE + c * CSTRIDE + ch * 64);
#pragma unroll
    for (int e = 0; e < 32; ++e) dst[e] = (__bf16)src[e];
  }
  __syncthreads();

  // per-gate LDS fragment base addresses (lane-dependent, loop-invariant)
  const char* wl[4];
  float bias[4];
#pragma unroll
  for (int g = 0; g < 4; ++g) {
    wl[g] = smem + g * GSTRIDE + l15 * CSTRIDE + klB;
    bias[g] = bg[g][j * 16 + l15];
  }

  const unsigned axrowB = (unsigned)arow * (S_ * E_ * 2) + klB;  // x row base
  const unsigned aaoffB = (unsigned)arow * (H_ * 2) + klB;       // a row base

  // C/D tile mapping: element r of lane l -> batch row, hidden col
  const unsigned obase = (unsigned)(mt * 16 + ((lane >> 4) << 3)) * H_ +
                         (unsigned)(j * 16 + l15);

  v8f cstate = splat8(0.0f);            // cell state lives in VGPRs
  const __bf16* acur = abuf;            // a_t   (read)
  __bf16*       anxt = abuf + B_ * H_;  // a_t+1 (write)

  for (int t = 0; t < S_; ++t) {
    v8f accf = splat8(bias[0]);
    v8f acci = splat8(bias[1]);
    v8f accc = splat8(bias[2]);
    v8f acco = splat8(bias[3]);

    const unsigned axoffB = axrowB + (unsigned)t * (E_ * 2);
    if (t + 1 < S_)
      __builtin_prefetch((const char*)xemb + axoffB + E_ * 2, 0, 1);

    // A fragment for chunk kc (kc is compile-time after unroll)
    auto ldA = [&](int kc) -> v16bf {
      return (kc < 16) ? ldfrag_g(xemb, axoffB, kc * 64)
                       : ldfrag_g(acur, aaoffB, kc * 64 - E_ * 2);
    };

    // ---- fully unrolled K loop; A loads pipelined APIPE chunks ahead ----
    v16bf apipe[APIPE];
#pragma unroll
    for (int p = 0; p < APIPE; ++p) apipe[p] = ldA(p);

#pragma unroll
    for (int kc = 0; kc < KCH; ++kc) {
      const int k0 = kc * 64;  // constant chunk byte offset
      const v16bf a = apipe[kc % APIPE];
      if (kc + APIPE < KCH) apipe[kc % APIPE] = ldA(kc + APIPE);
      v16bf f0 = ldfrag_l(wl[0], k0);
      v16bf f1 = ldfrag_l(wl[1], k0);
      v16bf f2 = ldfrag_l(wl[2], k0);
      v16bf f3 = ldfrag_l(wl[3], k0);
      wmma4(a, f0, f1, f2, f3, accf, acci, accc, acco);
    }

    // gate nonlinearities + state update (all lane-local on C-tile layout)
#pragma unroll
    for (int r = 0; r < 8; ++r) {
      const float fg = fast_sigmoid(accf[r]);
      const float ig = fast_sigmoid(acci[r]);
      const float cg = fast_tanh(accc[r]);
      const float og = fast_sigmoid(acco[r]);
      const float c  = fg * cstate[r] + ig * cg;
      cstate[r] = c;
      const float a = fast_tanh(c) * og;
      anxt[obase + (unsigned)r * H_] = (__bf16)a;
      if (t == S_ - 1) afin[obase + (unsigned)r * H_] = a;
    }

    // device-wide step barrier (monotonic counter, agent scope)
    __threadfence();
    __syncthreads();
    if (threadIdx.x == 0) {
      atomicAdd(cnt, 1u);
      const unsigned target = (unsigned)(NWG * (t + 1));
      while (__hip_atomic_load(cnt, __ATOMIC_RELAXED,
                               __HIP_MEMORY_SCOPE_AGENT) < target)
        __builtin_amdgcn_s_sleep(1);
    }
    __syncthreads();
    __threadfence();

    // swap a buffers
    const __bf16* tmp = acur;
    acur = anxt;
    anxt = (__bf16*)tmp;
  }
}

// ------------------------------- classifier --------------------------------

__global__ void classifier_loss(const float* __restrict__ afin,
                                const float* __restrict__ Wv,
                                const float* __restrict__ bv,
                                const int* __restrict__ label,
                                float* __restrict__ out) {
  __shared__ float sm[B_];
  const int b = threadIdx.x;  // blockDim.x == B_
  if (b < B_) {
    float logits[C_];
    const float* a = afin + (size_t)b * H_;
#pragma unroll 1
    for (int c = 0; c < C_; ++c) {
      const float* w = Wv + (size_t)c * H_;
      float d = bv[c];
      for (int h = 0; h < H_; ++h) d = fmaf(a[h], w[h], d);
      logits[c] = d;
    }
    float mx = logits[0];
    for (int c = 1; c < C_; ++c) mx = fmaxf(mx, logits[c]);
    float se = 0.f;
    for (int c = 0; c < C_; ++c) se += __expf(logits[c] - mx);
    const float lse = mx + __logf(se);
    sm[b] = logits[label[b]] - lse;
  }
  __syncthreads();
  if (threadIdx.x == 0) {
    float s = 0.f;
    for (int i = 0; i < B_; ++i) s += sm[i];
    out[0] = -s / (float)B_;
  }
}

// ------------------------------- launcher ----------------------------------

extern "C" void kernel_launch(void* const* d_in, const int* in_sizes, int n_in,
                              void* d_out, int out_size, void* d_ws, size_t ws_size,
                              hipStream_t stream) {
  const int*   x     = (const int*)d_in[0];
  const int*   label = (const int*)d_in[1];
  const float* emb   = (const float*)d_in[2];
  const float* Wf    = (const float*)d_in[3];
  const float* bf    = (const float*)d_in[4];
  const float* Wi    = (const float*)d_in[5];
  const float* bi    = (const float*)d_in[6];
  const float* Wc    = (const float*)d_in[7];
  const float* bc    = (const float*)d_in[8];
  const float* Wo    = (const float*)d_in[9];
  const float* bo    = (const float*)d_in[10];
  const float* Wv    = (const float*)d_in[11];
  const float* bv    = (const float*)d_in[12];

  char* ws = (char*)d_ws;
  __bf16*   xemb = (__bf16*)(ws + XEMB_OFF);
  __bf16*   abuf = (__bf16*)(ws + ABUF_OFF);
  float*    afin = (float*)(ws + AFIN_OFF);
  unsigned* cnt  = (unsigned*)(ws + CNT_OFF);

  gather_embed<<<B_ * S_, 256, 0, stream>>>(x, emb, xemb);
  init_state<<<64, 256, 0, stream>>>(abuf, cnt);
  lstm_recurrent<<<NWG, 128, LDSZ, stream>>>(xemb, Wf, Wi, Wc, Wo, bf, bi, bc,
                                             bo, abuf, afin, cnt);
  classifier_loss<<<1, B_, 0, stream>>>(afin, Wv, bv, label, (float*)d_out);
}
